// GenRec_18038862643865
// MI455X (gfx1250) — matrix operand order repaired and checked by reading, not compiled, not run on hardware
//
#include <hip/hip_runtime.h>
#include <hip/hip_bf16.h>

typedef __attribute__((ext_vector_type(16))) _Float16 v16h;
typedef __attribute__((ext_vector_type(8)))  float    v8f;

#define BB   1024
#define SQ   50
#define KK   30
#define DD   128
#define HH   4
#define DH   32
#define LL   2
#define FFD  2048
#define MM   (BB*SQ)   // 51200 rows

// ---------------------------------------------------------------------------
// WMMA fragment helpers (gfx1250 wave32 layouts, CDNA5 ISA 7.12.2)
// ---------------------------------------------------------------------------
// A: 16x32 f16.  lane&15 = M row; lane>>4 selects K halves {0..7,16..23} vs
// {8..15,24..31}.  VGPR j: j<4 -> K = 2j(+half*8), j>=4 -> K = 16+2(j-4)(+half*8).
__device__ inline v16h load_a_frag(const _Float16* __restrict__ A, int lda,
                                   int mBase, int kBase, int lane) {
    const _Float16* row = A + (size_t)(mBase + (lane & 15)) * lda + kBase;
    int hb = (lane >> 4) * 8;
    v16h a;
#pragma unroll
    for (int j = 0; j < 8; ++j) {
        int kb = ((j < 4) ? (2 * j) : (16 + 2 * (j - 4))) + hb;
        a[2 * j]     = row[kb];
        a[2 * j + 1] = row[kb + 1];
    }
    return a;
}

// B: 32x16 f16 (K x N), sourced from row-major W[N][K] so B[k][n] = W[n][k].
// lane&15 = N col; lane>>4 selects K halves {0..7,16..23} vs {8..15,24..31} (same as A).
__device__ inline v16h load_b_frag(const _Float16* __restrict__ W, int ldw,
                                   int nBase, int kBase, int lane) {
    typedef __attribute__((ext_vector_type(8))) _Float16 v8h_;
    const _Float16* p = W + (size_t)(nBase + (lane & 15)) * ldw + kBase + (lane >> 4) * 8;
    v8h_ lo = *(const v8h_*)p;
    v8h_ hi = *(const v8h_*)(p + 16);
    return __builtin_shufflevector(lo, hi, 0,1,2,3,4,5,6,7,8,9,10,11,12,13,14,15);
}

__device__ inline v8f wmma_f16(v16h a, v16h b, v8f c) {
    return __builtin_amdgcn_wmma_f32_16x16x32_f16(false, a, false, b, (short)0, c,
                                                  false, false);
}

// D/C: 16x16 f32. c[r] -> row (r + 8*(lane>>4)), col (lane&15).

// ---------------------------------------------------------------------------
// f32 -> f16 convert
// ---------------------------------------------------------------------------
__global__ void cvt_f32_f16(const float* __restrict__ src, _Float16* __restrict__ dst, int n) {
    int i = blockIdx.x * blockDim.x + threadIdx.x;
    if (i < n) dst[i] = (_Float16)src[i];
}

// ---------------------------------------------------------------------------
// Ragged basket pooling + position embedding + basket embedding
// one block per (b,s) row, 128 threads = D lanes.  Item-id stream is
// wave-uniform -> coalesced gathers along D (table is L2-resident: 51MB).
// ---------------------------------------------------------------------------
__global__ __launch_bounds__(128) void pool_kernel(
        const float* __restrict__ item_emb, const float* __restrict__ pos_emb,
        const float* __restrict__ basket_emb, const int* __restrict__ seq,
        float* __restrict__ x) {
    int row = blockIdx.x;          // b*SQ + s
    int s   = row % SQ;
    int d   = threadIdx.x;
    const int* bs = seq + (size_t)row * KK;
    float acc = 0.f;
    int   cnt = 0;
    for (int k = 0; k < KK; ++k) {
        int id = bs[k];
        if (id > 0) {
            acc += item_emb[(size_t)id * DD + d];
            int r = cnt;                       // rank = cumsum(valid)-1
            if (r > 0) acc += pos_emb[(size_t)r * DD + d];  // pos 0 is padding row
            cnt++;
        }
    }
    float pooled = (cnt > 0) ? acc / (float)cnt : 0.f;
    float be = (s > 0) ? basket_emb[(size_t)s * DD + d] : 0.f;   // basket_ids==s
    x[(size_t)row * DD + d] = pooled + be;
}

// ---------------------------------------------------------------------------
// Generic WMMA GEMM: C[M,N] = A_h[M,K] * W_h[N,K]^T + bias[N]  (opt. ReLU)
// block = 4 waves; block tile 128(M) x 64(N); each wave: 32 x 64 strip
// (2 M-frags x 4 N-frags = 8 WMMA per K-step from 6 fragment loads).
// Ping-pong K pipeline (unrolled x2, Kt % 64 == 0): buffer P is loaded while
// buffer Q feeds the WMMA burst -- no rotation copies, loads overlap issue.
// ---------------------------------------------------------------------------
__global__ __launch_bounds__(128) void wmma_gemm(
        const _Float16* __restrict__ A, const _Float16* __restrict__ W,
        const float* __restrict__ bias, float* __restrict__ C,
        int M, int N, int Kt, int relu) {
    int wave = threadIdx.x >> 5;
    int lane = threadIdx.x & 31;
    int mBase = blockIdx.y * 128 + wave * 32;
    int nBase = blockIdx.x * 64;
    v8f acc[2][4] = {};
    v16h aP[2], bP[4], aQ[2], bQ[4];
#pragma unroll
    for (int mt = 0; mt < 2; ++mt) aP[mt] = load_a_frag(A, Kt, mBase + mt * 16, 0, lane);
#pragma unroll
    for (int t = 0; t < 4; ++t) bP[t] = load_b_frag(W, Kt, nBase + t * 16, 0, lane);
    for (int k = 0; k < Kt; k += 64) {
        // prefetch Q @ k+32 (always valid: Kt is a multiple of 64)
#pragma unroll
        for (int mt = 0; mt < 2; ++mt) aQ[mt] = load_a_frag(A, Kt, mBase + mt * 16, k + 32, lane);
#pragma unroll
        for (int t = 0; t < 4; ++t) bQ[t] = load_b_frag(W, Kt, nBase + t * 16, k + 32, lane);
        // consume P
#pragma unroll
        for (int mt = 0; mt < 2; ++mt)
#pragma unroll
            for (int t = 0; t < 4; ++t)
                acc[mt][t] = wmma_f16(aP[mt], bP[t], acc[mt][t]);
        // prefetch P @ k+64 (skip past end)
        if (k + 64 < Kt) {
#pragma unroll
            for (int mt = 0; mt < 2; ++mt) aP[mt] = load_a_frag(A, Kt, mBase + mt * 16, k + 64, lane);
#pragma unroll
            for (int t = 0; t < 4; ++t) bP[t] = load_b_frag(W, Kt, nBase + t * 16, k + 64, lane);
        }
        // consume Q
#pragma unroll
        for (int mt = 0; mt < 2; ++mt)
#pragma unroll
            for (int t = 0; t < 4; ++t)
                acc[mt][t] = wmma_f16(aQ[mt], bQ[t], acc[mt][t]);
    }
    int hb = lane >> 4, n0 = lane & 15;
#pragma unroll
    for (int mt = 0; mt < 2; ++mt) {
#pragma unroll
        for (int t = 0; t < 4; ++t) {
            int n = nBase + t * 16 + n0;
            float bb = bias[n];
#pragma unroll
            for (int r = 0; r < 8; ++r) {
                float v = acc[mt][t][r] + bb;
                if (relu) v = fmaxf(v, 0.f);
                C[(size_t)(mBase + mt * 16 + r + 8 * hb) * N + n] = v;
            }
        }
    }
}

// ---------------------------------------------------------------------------
// Attention: one block per (b, head); 64 threads; K/V staged in LDS.
// qkv layout per row: [q(128) | k(128) | v(128)], head h = cols h*32..h*32+31.
// ---------------------------------------------------------------------------
__global__ __launch_bounds__(64) void attention_kernel(
        const float* __restrict__ qkv, const int* __restrict__ lengths,
        float* __restrict__ O) {
    __shared__ float kbuf[SQ * DH];
    __shared__ float vbuf[SQ * DH];
    int b = blockIdx.x, h = blockIdx.y;
    int tid = threadIdx.x;
    const float* base = qkv + (size_t)b * SQ * (3 * DD);
    for (int i = tid; i < SQ * DH; i += 64) {
        int r = i >> 5, d = i & 31;
        kbuf[i] = base[r * (3 * DD) + DD + h * DH + d];
        vbuf[i] = base[r * (3 * DD) + 2 * DD + h * DH + d];
    }
    __syncthreads();
    if (tid < SQ) {
        float q[DH];
#pragma unroll
        for (int d = 0; d < DH; ++d) q[d] = base[tid * (3 * DD) + h * DH + d];
        int len = lengths[b];
        float sc[SQ];
        float mx = -1e30f;
        for (int k = 0; k < SQ; ++k) {
            float s = 0.f;
#pragma unroll
            for (int d = 0; d < DH; ++d) s += q[d] * kbuf[k * DH + d];
            s *= 0.17677669529663687f;            // 1/sqrt(32)
            if (k >= len) s = -1e9f;
            sc[k] = s;
            mx = fmaxf(mx, s);
        }
        float sum = 0.f;
        for (int k = 0; k < SQ; ++k) { float e = __expf(sc[k] - mx); sc[k] = e; sum += e; }
        float inv = 1.f / sum;
        float o[DH];
#pragma unroll
        for (int d = 0; d < DH; ++d) o[d] = 0.f;
        for (int k = 0; k < SQ; ++k) {
            float w = sc[k] * inv;
#pragma unroll
            for (int d = 0; d < DH; ++d) o[d] += w * vbuf[k * DH + d];
        }
        float* op = O + ((size_t)b * SQ + tid) * DD + h * DH;
#pragma unroll
        for (int d = 0; d < DH; ++d) op[d] = o[d];
    }
}

// ---------------------------------------------------------------------------
// x = LayerNorm(x + y) * g + b      (one block per row, 128 threads)
// ---------------------------------------------------------------------------
__global__ __launch_bounds__(128) void add_ln_kernel(
        float* __restrict__ x, const float* __restrict__ y,
        const float* __restrict__ g, const float* __restrict__ bt) {
    __shared__ float red[DD];
    int row = blockIdx.x, d = threadIdx.x;
    size_t idx = (size_t)row * DD + d;
    float v = x[idx] + y[idx];
    red[d] = v; __syncthreads();
    for (int s = 64; s > 0; s >>= 1) { if (d < s) red[d] += red[d + s]; __syncthreads(); }
    float mean = red[0] * (1.f / DD);
    __syncthreads();
    float dv = v - mean;
    red[d] = dv * dv; __syncthreads();
    for (int s = 64; s > 0; s >>= 1) { if (d < s) red[d] += red[d + s]; __syncthreads(); }
    float var = red[0] * (1.f / DD);
    x[idx] = dv * rsqrtf(var + 1e-5f) * g[d] + bt[d];
}

// ---------------------------------------------------------------------------
// Fused FF: F = relu(Xh @ W1^T + b1) @ W2^T + b2.
// 32-row tile per block; the 32x2048 ReLU hidden lives in 128KB LDS (f16) --
// fine on a 320KB WGP, and avoids a 210MB global intermediate.
// Stage 1 reuses every W1 fragment across two M-frags (2x compute/byte);
// stage 2 is a ping-pong pipelined 2x2 register block over K=2048.
// ---------------------------------------------------------------------------
__global__ __launch_bounds__(128) void ff_fused_kernel(
        const _Float16* __restrict__ Xh,
        const _Float16* __restrict__ W1h, const float* __restrict__ b1v,
        const _Float16* __restrict__ W2h, const float* __restrict__ b2v,
        float* __restrict__ F) {
    extern __shared__ _Float16 hbuf[];      // 32 * 2048 f16 = 128 KB
    int wave = threadIdx.x >> 5;
    int lane = threadIdx.x & 31;
    int mBase = blockIdx.x * 32;
    int hb = lane >> 4, n0 = lane & 15;

    // preload the 32x128 A tile as 2x4 K-step fragments (reused across 32 N tiles)
    v16h afrag[2][4];
#pragma unroll
    for (int mt = 0; mt < 2; ++mt)
#pragma unroll
        for (int ks = 0; ks < 4; ++ks)
            afrag[mt][ks] = load_a_frag(Xh, DD, mBase + mt * 16, ks * 32, lane);

    // stage 1: hidden = relu(X @ W1^T + b1), wave w covers N in [w*512, w*512+512)
    for (int nt = 0; nt < 32; ++nt) {
        int n = wave * 512 + nt * 16;
        v8f acc0 = {}, acc1 = {};
#pragma unroll
        for (int ks = 0; ks < 4; ++ks) {
            v16h b = load_b_frag(W1h, DD, n, ks * 32, lane);
            acc0 = wmma_f16(afrag[0][ks], b, acc0);
            acc1 = wmma_f16(afrag[1][ks], b, acc1);
        }
        float bb = b1v[n + n0];
#pragma unroll
        for (int r = 0; r < 8; ++r) {
            hbuf[(size_t)(r + 8 * hb) * FFD + n + n0]      = (_Float16)fmaxf(acc0[r] + bb, 0.f);
            hbuf[(size_t)(16 + r + 8 * hb) * FFD + n + n0] = (_Float16)fmaxf(acc1[r] + bb, 0.f);
        }
    }
    __syncthreads();

    // stage 2: F = hidden @ W2^T + b2 ; wave handles N tiles {2w, 2w+1} x both M.
    // Ping-pong pipeline over the 64 K-steps (K = 2048).
    v8f acc2[2][2] = {};
    int nA = (wave * 2) * 16, nB2 = (wave * 2 + 1) * 16;
    v16h a0P, a1P, b0P, b1P, a0Q, a1Q, b0Q, b1Q;
    a0P = load_a_frag(hbuf, FFD, 0,  0, lane);
    a1P = load_a_frag(hbuf, FFD, 16, 0, lane);
    b0P = load_b_frag(W2h, FFD, nA,  0, lane);
    b1P = load_b_frag(W2h, FFD, nB2, 0, lane);
    for (int ks = 0; ks < 64; ks += 2) {
        int kQ = (ks + 1) * 32;
        a0Q = load_a_frag(hbuf, FFD, 0,  kQ, lane);
        a1Q = load_a_frag(hbuf, FFD, 16, kQ, lane);
        b0Q = load_b_frag(W2h, FFD, nA,  kQ, lane);
        b1Q = load_b_frag(W2h, FFD, nB2, kQ, lane);
        acc2[0][0] = wmma_f16(a0P, b0P, acc2[0][0]);
        acc2[0][1] = wmma_f16(a0P, b1P, acc2[0][1]);
        acc2[1][0] = wmma_f16(a1P, b0P, acc2[1][0]);
        acc2[1][1] = wmma_f16(a1P, b1P, acc2[1][1]);
        if (ks + 2 < 64) {
            int kP = (ks + 2) * 32;
            a0P = load_a_frag(hbuf, FFD, 0,  kP, lane);
            a1P = load_a_frag(hbuf, FFD, 16, kP, lane);
            b0P = load_b_frag(W2h, FFD, nA,  kP, lane);
            b1P = load_b_frag(W2h, FFD, nB2, kP, lane);
        }
        acc2[0][0] = wmma_f16(a0Q, b0Q, acc2[0][0]);
        acc2[0][1] = wmma_f16(a0Q, b1Q, acc2[0][1]);
        acc2[1][0] = wmma_f16(a1Q, b0Q, acc2[1][0]);
        acc2[1][1] = wmma_f16(a1Q, b1Q, acc2[1][1]);
    }
#pragma unroll
    for (int mt = 0; mt < 2; ++mt) {
#pragma unroll
        for (int j = 0; j < 2; ++j) {
            int n = (wave * 2 + j) * 16 + n0;
            float bb = b2v[n];
#pragma unroll
            for (int r = 0; r < 8; ++r)
                F[(size_t)(mBase + mt * 16 + r + 8 * hb) * DD + n] = acc2[mt][j][r] + bb;
        }
    }
}

// ---------------------------------------------------------------------------
// out = x masked past per-user length
// ---------------------------------------------------------------------------
__global__ __launch_bounds__(128) void mask_out_kernel(
        const float* __restrict__ x, const int* __restrict__ lengths,
        float* __restrict__ out) {
    int row = blockIdx.x;
    int b = row / SQ, s = row % SQ;
    int d = threadIdx.x;
    size_t idx = (size_t)row * DD + d;
    out[idx] = (s < lengths[b]) ? x[idx] : 0.f;
}

// ---------------------------------------------------------------------------
extern "C" void kernel_launch(void* const* d_in, const int* in_sizes, int n_in,
                              void* d_out, int out_size, void* d_ws, size_t ws_size,
                              hipStream_t stream) {
    const float* item_emb   = (const float*)d_in[0];
    const float* pos_emb    = (const float*)d_in[1];
    const float* basket_emb = (const float*)d_in[2];
    const float* Wqkv  = (const float*)d_in[3];
    const float* bqkv  = (const float*)d_in[4];
    const float* Wo    = (const float*)d_in[5];
    const float* bo    = (const float*)d_in[6];
    const float* W1    = (const float*)d_in[7];
    const float* b1    = (const float*)d_in[8];
    const float* W2    = (const float*)d_in[9];
    const float* b2    = (const float*)d_in[10];
    const float* ln1_s = (const float*)d_in[11];
    const float* ln1_b = (const float*)d_in[12];
    const float* ln2_s = (const float*)d_in[13];
    const float* ln2_b = (const float*)d_in[14];
    const int* basket_seq = (const int*)d_in[15];
    // d_in[16] position_ids: unused (overwritten in reference)
    // d_in[17] basket_ids == arange(S): handled analytically
    // d_in[18] batch_size scalar: compile-time constant
    const int* lengths = (const int*)d_in[19];
    (void)in_sizes; (void)n_in; (void)out_size; (void)ws_size;

    // allow 128KB dynamic LDS for the fused FF kernel (WGP has 320KB)
    (void)hipFuncSetAttribute((const void*)ff_fused_kernel,
                              hipFuncAttributeMaxDynamicSharedMemorySize,
                              32 * FFD * (int)sizeof(_Float16));

    // workspace carve-up (all chunk sizes are multiples of 256B)
    char* p = (char*)d_ws;
    float* x    = (float*)p;      p += (size_t)MM * DD * 4;       // 26.2 MB
    float* qkv  = (float*)p;      p += (size_t)MM * 3 * DD * 4;   // 78.6 MB
    float* obuf = (float*)p;      p += (size_t)MM * DD * 4;       // 26.2 MB
    float* tmp  = (float*)p;      p += (size_t)MM * DD * 4;       // 26.2 MB
    _Float16* xh = (_Float16*)p;  p += (size_t)MM * DD * 2;       // 13.1 MB
    _Float16* oh = (_Float16*)p;  p += (size_t)MM * DD * 2;       // 13.1 MB
    _Float16* wqkv_h = (_Float16*)p; p += (size_t)LL * 3 * DD * DD * 2;
    _Float16* wo_h   = (_Float16*)p; p += (size_t)LL * DD * DD * 2;
    _Float16* w1_h   = (_Float16*)p; p += (size_t)LL * FFD * DD * 2;
    _Float16* w2_h   = (_Float16*)p; p += (size_t)LL * DD * FFD * 2;

    // weights -> f16 (once per launch; 2.25 MB total, then L2-resident)
    {
        int n;
        n = LL * 3 * DD * DD; cvt_f32_f16<<<(n + 255) / 256, 256, 0, stream>>>(Wqkv, wqkv_h, n);
        n = LL * DD * DD;     cvt_f32_f16<<<(n + 255) / 256, 256, 0, stream>>>(Wo, wo_h, n);
        n = LL * FFD * DD;    cvt_f32_f16<<<(n + 255) / 256, 256, 0, stream>>>(W1, w1_h, n);
        n = LL * DD * FFD;    cvt_f32_f16<<<(n + 255) / 256, 256, 0, stream>>>(W2, w2_h, n);
    }

    // ragged pooling -> x [MM, 128]
    pool_kernel<<<MM, 128, 0, stream>>>(item_emb, pos_emb, basket_emb, basket_seq, x);

    const int nAct = MM * DD;
    for (int l = 0; l < LL; ++l) {
        // QKV projection
        cvt_f32_f16<<<(nAct + 255) / 256, 256, 0, stream>>>(x, xh, nAct);
        wmma_gemm<<<dim3((3 * DD) / 64, MM / 128), 128, 0, stream>>>(
            xh, wqkv_h + (size_t)l * 3 * DD * DD, bqkv + (size_t)l * 3 * DD,
            qkv, MM, 3 * DD, DD, 0);
        // attention
        attention_kernel<<<dim3(BB, HH), 64, 0, stream>>>(qkv, lengths, obuf);
        // output projection
        cvt_f32_f16<<<(nAct + 255) / 256, 256, 0, stream>>>(obuf, oh, nAct);
        wmma_gemm<<<dim3(DD / 64, MM / 128), 128, 0, stream>>>(
            oh, wo_h + (size_t)l * DD * DD, bo + (size_t)l * DD,
            tmp, MM, DD, DD, 0);
        // residual + LN1
        add_ln_kernel<<<MM, 128, 0, stream>>>(x, tmp, ln1_s + (size_t)l * DD, ln1_b + (size_t)l * DD);
        // fused feed-forward
        cvt_f32_f16<<<(nAct + 255) / 256, 256, 0, stream>>>(x, xh, nAct);
        ff_fused_kernel<<<MM / 32, 128, 32 * FFD * sizeof(_Float16), stream>>>(
            xh, w1_h + (size_t)l * FFD * DD, b1 + (size_t)l * FFD,
            w2_h + (size_t)l * DD * FFD, b2 + (size_t)l * DD, tmp);
        // residual + LN2
        add_ln_kernel<<<MM, 128, 0, stream>>>(x, tmp, ln2_s + (size_t)l * DD, ln2_b + (size_t)l * DD);
    }

    // zero rows past per-user length, write final output
    mask_out_kernel<<<MM, 128, 0, stream>>>(x, lengths, (float*)d_out);
}
